// BambaMixerDecoderLayer_34892314313355
// MI455X (gfx1250) — compile-verified, matrix operand order checked by reading, not run on hardware
//
#include <hip/hip_runtime.h>
#include <hip/hip_bf16.h>
#include <stdint.h>

// ---------------- problem constants (match reference) ----------------
#define HID   2048
#define DI    4096
#define NH    64
#define PD    64          // P
#define NS    128         // N
#define KC    4
#define IM    8192
#define BB    2
#define LL    2048
#define ROWS  (BB*LL)                  // 4096
#define CONV  (DI + 2*NS)              // 4352
#define DPROJ (2*DI + 2*NS + NH)       // 8512
#define EPSN  1e-5f

typedef __bf16 bf16x16 __attribute__((ext_vector_type(16)));
typedef float  f32x8   __attribute__((ext_vector_type(8)));

__device__ __forceinline__ unsigned short f2bf(float f) {
  union { float f; unsigned int u; } c; c.f = f;
  unsigned int u = c.u;
  unsigned int r = (u + 0x7FFFu + ((u >> 16) & 1u)) >> 16;   // RNE
  if ((u & 0x7F800000u) == 0x7F800000u && (u & 0x007FFFFFu)) r = (u >> 16) | 0x0040u;
  return (unsigned short)r;
}

__device__ __forceinline__ float siluf(float x) { return x / (1.f + __expf(-x)); }

__device__ float block_reduce_sum256(float v) {
  __shared__ float red[8];
  int lane = threadIdx.x & 31, w = threadIdx.x >> 5;
  #pragma unroll
  for (int o = 16; o; o >>= 1) v += __shfl_xor(v, o, 32);
  if (lane == 0) red[w] = v;
  __syncthreads();
  float r = 0.f;
  if (threadIdx.x < 8) r = red[threadIdx.x];
  if (w == 0) {
    #pragma unroll
    for (int o = 4; o; o >>= 1) r += __shfl_xor(r, o, 32);
  }
  if (threadIdx.x == 0) red[0] = r;
  __syncthreads();
  float out = red[0];
  __syncthreads();
  return out;
}

// ---------------- f32 -> bf16 conversion (weights; NT reads) ----------------
__global__ void cvt_bf16_kernel(const float* __restrict__ src,
                                unsigned short* __restrict__ dst, long n) {
  long i = (long)blockIdx.x * blockDim.x + threadIdx.x;
  long stride = (long)gridDim.x * blockDim.x;
  for (; i < n; i += stride) dst[i] = f2bf(__builtin_nontemporal_load(src + i));
}

// ---------------- residual add + RMSNorm -> bf16 (and optional f32 resid out) ----
__global__ __launch_bounds__(256)
void addnorm_kernel(const float* __restrict__ a, const float* __restrict__ b,
                    const float* __restrict__ w, float* __restrict__ residout,
                    unsigned short* __restrict__ bfout) {
  const int VPT = HID / 256;
  long row = blockIdx.x;
  const float* ar = a + row * HID;
  const float* br = b ? b + row * HID : nullptr;
  float v[VPT]; float ss = 0.f;
  #pragma unroll
  for (int i = 0; i < VPT; i++) {
    int col = threadIdx.x + i * 256;
    float x = ar[col] + (br ? br[col] : 0.f);
    v[i] = x; ss += x * x;
  }
  ss = block_reduce_sum256(ss);
  float sc = rsqrtf(ss / (float)HID + EPSN);
  #pragma unroll
  for (int i = 0; i < VPT; i++) {
    int col = threadIdx.x + i * 256;
    if (residout) residout[row * HID + col] = v[i];
    bfout[row * HID + col] = f2bf(v[i] * sc * w[col]);
  }
}

// ---------------- y*silu(z) + mixer RMSNorm -> bf16 ----------------
__global__ __launch_bounds__(256)
void gatenorm_kernel(const float* __restrict__ y, const float* __restrict__ proj,
                     const float* __restrict__ w, unsigned short* __restrict__ out) {
  const int VPT = DI / 256;
  long row = blockIdx.x;
  const float* yr = y + row * DI;
  const float* zr = proj + row * (long)DPROJ;   // z = proj[..., :DI]
  float g[VPT]; float ss = 0.f;
  #pragma unroll
  for (int i = 0; i < VPT; i++) {
    int col = threadIdx.x + i * 256;
    float gv = yr[col] * siluf(zr[col]);
    g[i] = gv; ss += gv * gv;
  }
  ss = block_reduce_sum256(ss);
  float sc = rsqrtf(ss / (float)DI + EPSN);
  #pragma unroll
  for (int i = 0; i < VPT; i++) {
    int col = threadIdx.x + i * 256;
    out[row * DI + col] = f2bf(g[i] * sc * w[col]);
  }
}

// ---------------- depthwise causal conv (K=4) + SiLU ----------------
__global__ void conv_silu_kernel(const float* __restrict__ proj,
                                 const float* __restrict__ cw,
                                 const float* __restrict__ cb,
                                 float* __restrict__ out) {
  long total = (long)ROWS * CONV;
  long i = (long)blockIdx.x * blockDim.x + threadIdx.x;
  long stride = (long)gridDim.x * blockDim.x;
  for (; i < total; i += stride) {
    int c = (int)(i % CONV);
    long r = i / CONV;
    int l = (int)(r % LL);
    long b = r / LL;
    float acc = cb[c];
    #pragma unroll
    for (int k = 0; k < KC; k++) {
      int lk = l + k - (KC - 1);
      if (lk >= 0)
        acc += proj[(b * LL + lk) * (long)DPROJ + DI + c] * cw[c * KC + k];
    }
    out[i] = siluf(acc);
  }
}

// ---------------- dt = softplus(raw + bias); dA = exp(dt * -exp(A_log)) --------
__global__ void dt_kernel(const float* __restrict__ proj,
                          const float* __restrict__ dt_bias,
                          const float* __restrict__ A_log,
                          float* __restrict__ dtv, float* __restrict__ dAv) {
  int i = blockIdx.x * blockDim.x + threadIdx.x;
  if (i >= ROWS * NH) return;
  int h = i % NH; long r = i / NH;
  float raw = proj[r * (long)DPROJ + DI + CONV + h] + dt_bias[h];
  float dt = raw > 20.f ? raw : log1pf(__expf(raw));
  float a = -__expf(A_log[h]);
  dtv[i] = dt;
  dAv[i] = __expf(dt * a);
}

// ---------------- recurrent selective scan: one block per (b,h) ----------------
// state h[p][n] lives in registers: thread t owns p = t>>2, n = (t&3)*32 .. +31
__global__ __launch_bounds__(256)
void scan_kernel(const float* __restrict__ xbc, const float* __restrict__ dtv,
                 const float* __restrict__ dAv, const float* __restrict__ Dv,
                 float* __restrict__ y) {
  int b = blockIdx.x / NH;
  int h = blockIdx.x % NH;
  int t = threadIdx.x;
  int p = t >> 2;
  int sub = t & 3;
  int n0 = sub * 32;
  float Dh = Dv[h];

  float hs[32];
  #pragma unroll
  for (int j = 0; j < 32; j++) hs[j] = 0.f;

  __shared__ float sB[NS], sC[NS], sX[PD];

  for (int l = 0; l < LL; l++) {
    long base = (long)b * LL + l;
    const float* row = xbc + base * CONV;
    if (t < NS)            sB[t]       = row[DI + t];
    else                   sC[t - NS]  = row[DI + NS + (t - NS)];
    if (t < PD)            sX[t]       = row[h * PD + t];
    float dAc = dAv[base * NH + h];
    float dtc = dtv[base * NH + h];
    __syncthreads();
    float xv = sX[p];
    float coef = dtc * xv;
    float acc = 0.f;
    #pragma unroll
    for (int j = 0; j < 32; j++) {
      float hv = dAc * hs[j] + coef * sB[n0 + j];
      hs[j] = hv;
      acc += hv * sC[n0 + j];
    }
    acc += __shfl_xor(acc, 1, 32);
    acc += __shfl_xor(acc, 2, 32);
    if (sub == 0) y[base * DI + h * PD + p] = acc + Dh * xv;
    __syncthreads();
  }
}

// ---------------- WMMA bf16 GEMM tiles ----------------
#define BM 128
#define BN 64
#define BN2 128
#define BK 32
#define LDSS 40   // BK + 8 pad (bf16 elements); rows 80B apart -> 16B aligned

// C[M,E] = A[M,K] * W[E,K]^T (+addsrc), BN=64 (E multiple of 64)
__global__ __launch_bounds__(256)
void wmma_gemm_kernel(const unsigned short* __restrict__ A,
                      const unsigned short* __restrict__ W,
                      float* __restrict__ C, const float* __restrict__ addsrc,
                      int M, int Kd, int E) {
  __shared__ unsigned short sA[BM * LDSS];
  __shared__ unsigned short sW[BN * LDSS];
  int t = threadIdx.x;
  int lane = t & 31, wave = t >> 5;
  int wm = wave >> 1, wn = wave & 1;          // 4x2 wave grid, 32x32 per wave
  long rowBase = (long)blockIdx.y * BM;
  long colBase = (long)blockIdx.x * BN;

  f32x8 acc[2][2];
  #pragma unroll
  for (int mi = 0; mi < 2; mi++)
    #pragma unroll
    for (int ni = 0; ni < 2; ni++)
      #pragma unroll
      for (int r = 0; r < 8; r++) acc[mi][ni][r] = 0.f;

  int aRow = t >> 1, aCol = (t & 1) * 16;     // 128x32 A tile, 16 bf16/thread
  int wRow = t >> 2, wCol = (t & 3) * 8;      // 64x32 W tile,   8 bf16/thread
  int fr = lane & 15, kk = (lane >> 4) * 16;

  for (int k0 = 0; k0 < Kd; k0 += BK) {
    const uint4* ag = (const uint4*)(A + (rowBase + aRow) * (long)Kd + k0 + aCol);
    uint4 a0 = ag[0], a1 = ag[1];
    const uint4* wg = (const uint4*)(W + (colBase + wRow) * (long)Kd + k0 + wCol);
    uint4 w0 = wg[0];
    if (k0 + BK < Kd) {                        // -> global_prefetch_b8
      __builtin_prefetch(A + (rowBase + aRow) * (long)Kd + k0 + BK + aCol, 0, 1);
      __builtin_prefetch(W + (colBase + wRow) * (long)Kd + k0 + BK + wCol, 0, 1);
    }
    __syncthreads();
    *(uint4*)&sA[aRow * LDSS + aCol]     = a0;
    *(uint4*)&sA[aRow * LDSS + aCol + 8] = a1;
    *(uint4*)&sW[wRow * LDSS + wCol]     = w0;
    __syncthreads();

    union FB { uint4 u[2]; bf16x16 v; };
    FB af[2], bf[2];
    #pragma unroll
    for (int i = 0; i < 2; i++) {
      const uint4* pa = (const uint4*)&sA[(wm * 32 + i * 16 + fr) * LDSS + kk];
      af[i].u[0] = pa[0]; af[i].u[1] = pa[1];
      const uint4* pb = (const uint4*)&sW[(wn * 32 + i * 16 + fr) * LDSS + kk];
      bf[i].u[0] = pb[0]; bf[i].u[1] = pb[1];
    }
    #pragma unroll
    for (int mi = 0; mi < 2; mi++)
      #pragma unroll
      for (int ni = 0; ni < 2; ni++)
        acc[mi][ni] = __builtin_amdgcn_wmma_f32_16x16x32_bf16(
            false, af[mi].v, false, bf[ni].v, (short)0, acc[mi][ni], false, false);
  }

  int rsel = (lane >> 4) * 8, cc = lane & 15;
  #pragma unroll
  for (int mi = 0; mi < 2; mi++)
    #pragma unroll
    for (int ni = 0; ni < 2; ni++) {
      long col = colBase + wn * 32 + ni * 16 + cc;
      #pragma unroll
      for (int r = 0; r < 8; r++) {
        long row = rowBase + wm * 32 + mi * 16 + rsel + r;
        long idx = row * (long)E + col;
        float v = acc[mi][ni][r];
        if (addsrc) v += addsrc[idx];
        C[idx] = v;
      }
    }
}

// C[M,E] = A[M,K] * W[E,K]^T (+addsrc), BN=128, 32x64 wave tile (8 WMMA/iter)
__global__ __launch_bounds__(256)
void wmma_gemm128_kernel(const unsigned short* __restrict__ A,
                         const unsigned short* __restrict__ W,
                         float* __restrict__ C, const float* __restrict__ addsrc,
                         int M, int Kd, int E) {
  __shared__ unsigned short sA[BM * LDSS];
  __shared__ unsigned short sW[BN2 * LDSS];
  int t = threadIdx.x;
  int lane = t & 31, wave = t >> 5;
  int wm = wave >> 1, wn = wave & 1;          // wave tile: 32 rows x 64 cols
  long rowBase = (long)blockIdx.y * BM;
  long colBase = (long)blockIdx.x * BN2;

  f32x8 acc[2][4];
  #pragma unroll
  for (int mi = 0; mi < 2; mi++)
    #pragma unroll
    for (int ni = 0; ni < 4; ni++)
      #pragma unroll
      for (int r = 0; r < 8; r++) acc[mi][ni][r] = 0.f;

  int aRow = t >> 1, aCol = (t & 1) * 16;     // 128x32 A tile, 16 bf16/thread
  int wRow = t >> 1, wCol = (t & 1) * 16;     // 128x32 W tile, 16 bf16/thread
  int fr = lane & 15, kk = (lane >> 4) * 16;

  for (int k0 = 0; k0 < Kd; k0 += BK) {
    const uint4* ag = (const uint4*)(A + (rowBase + aRow) * (long)Kd + k0 + aCol);
    uint4 a0 = ag[0], a1 = ag[1];
    const uint4* wg = (const uint4*)(W + (colBase + wRow) * (long)Kd + k0 + wCol);
    uint4 w0 = wg[0], w1 = wg[1];
    if (k0 + BK < Kd) {
      __builtin_prefetch(A + (rowBase + aRow) * (long)Kd + k0 + BK + aCol, 0, 1);
      __builtin_prefetch(W + (colBase + wRow) * (long)Kd + k0 + BK + wCol, 0, 1);
    }
    __syncthreads();
    *(uint4*)&sA[aRow * LDSS + aCol]     = a0;
    *(uint4*)&sA[aRow * LDSS + aCol + 8] = a1;
    *(uint4*)&sW[wRow * LDSS + wCol]     = w0;
    *(uint4*)&sW[wRow * LDSS + wCol + 8] = w1;
    __syncthreads();

    union FB { uint4 u[2]; bf16x16 v; };
    FB af[2], bf[4];
    #pragma unroll
    for (int i = 0; i < 2; i++) {
      const uint4* pa = (const uint4*)&sA[(wm * 32 + i * 16 + fr) * LDSS + kk];
      af[i].u[0] = pa[0]; af[i].u[1] = pa[1];
    }
    #pragma unroll
    for (int j = 0; j < 4; j++) {
      const uint4* pb = (const uint4*)&sW[(wn * 64 + j * 16 + fr) * LDSS + kk];
      bf[j].u[0] = pb[0]; bf[j].u[1] = pb[1];
    }
    #pragma unroll
    for (int mi = 0; mi < 2; mi++)
      #pragma unroll
      for (int ni = 0; ni < 4; ni++)
        acc[mi][ni] = __builtin_amdgcn_wmma_f32_16x16x32_bf16(
            false, af[mi].v, false, bf[ni].v, (short)0, acc[mi][ni], false, false);
  }

  int rsel = (lane >> 4) * 8, cc = lane & 15;
  #pragma unroll
  for (int mi = 0; mi < 2; mi++)
    #pragma unroll
    for (int ni = 0; ni < 4; ni++) {
      long col = colBase + wn * 64 + ni * 16 + cc;
      #pragma unroll
      for (int r = 0; r < 8; r++) {
        long row = rowBase + wm * 32 + mi * 16 + rsel + r;
        long idx = row * (long)E + col;
        float v = acc[mi][ni][r];
        if (addsrc) v += addsrc[idx];
        C[idx] = v;
      }
    }
}

// ------- fused gate_up GEMM: act = silu(A*Wg^T) * (A*Wu^T), bf16 out ----------
__global__ __launch_bounds__(256)
void wmma_gateup_kernel(const unsigned short* __restrict__ A,
                        const unsigned short* __restrict__ Wg,
                        const unsigned short* __restrict__ Wu,
                        unsigned short* __restrict__ act, int M, int Kd) {
  __shared__ unsigned short sA[BM * LDSS];
  __shared__ unsigned short sG[BN * LDSS];
  __shared__ unsigned short sU[BN * LDSS];
  int t = threadIdx.x;
  int lane = t & 31, wave = t >> 5;
  int wm = wave >> 1, wn = wave & 1;
  long rowBase = (long)blockIdx.y * BM;
  long colBase = (long)blockIdx.x * BN;

  f32x8 accG[2][2], accU[2][2];
  #pragma unroll
  for (int mi = 0; mi < 2; mi++)
    #pragma unroll
    for (int ni = 0; ni < 2; ni++)
      #pragma unroll
      for (int r = 0; r < 8; r++) { accG[mi][ni][r] = 0.f; accU[mi][ni][r] = 0.f; }

  int aRow = t >> 1, aCol = (t & 1) * 16;
  int wRow = t >> 2, wCol = (t & 3) * 8;
  int fr = lane & 15, kk = (lane >> 4) * 16;

  for (int k0 = 0; k0 < Kd; k0 += BK) {
    const uint4* ag = (const uint4*)(A  + (rowBase + aRow) * (long)Kd + k0 + aCol);
    uint4 a0 = ag[0], a1 = ag[1];
    const uint4* gg = (const uint4*)(Wg + (colBase + wRow) * (long)Kd + k0 + wCol);
    uint4 g0 = gg[0];
    const uint4* ug = (const uint4*)(Wu + (colBase + wRow) * (long)Kd + k0 + wCol);
    uint4 u0 = ug[0];
    if (k0 + BK < Kd) {
      __builtin_prefetch(A  + (rowBase + aRow) * (long)Kd + k0 + BK + aCol, 0, 1);
      __builtin_prefetch(Wg + (colBase + wRow) * (long)Kd + k0 + BK + wCol, 0, 1);
      __builtin_prefetch(Wu + (colBase + wRow) * (long)Kd + k0 + BK + wCol, 0, 1);
    }
    __syncthreads();
    *(uint4*)&sA[aRow * LDSS + aCol]     = a0;
    *(uint4*)&sA[aRow * LDSS + aCol + 8] = a1;
    *(uint4*)&sG[wRow * LDSS + wCol]     = g0;
    *(uint4*)&sU[wRow * LDSS + wCol]     = u0;
    __syncthreads();

    union FB { uint4 u[2]; bf16x16 v; };
    FB af[2], gf[2], uf[2];
    #pragma unroll
    for (int i = 0; i < 2; i++) {
      const uint4* pa = (const uint4*)&sA[(wm * 32 + i * 16 + fr) * LDSS + kk];
      af[i].u[0] = pa[0]; af[i].u[1] = pa[1];
      const uint4* pg = (const uint4*)&sG[(wn * 32 + i * 16 + fr) * LDSS + kk];
      gf[i].u[0] = pg[0]; gf[i].u[1] = pg[1];
      const uint4* pu = (const uint4*)&sU[(wn * 32 + i * 16 + fr) * LDSS + kk];
      uf[i].u[0] = pu[0]; uf[i].u[1] = pu[1];
    }
    #pragma unroll
    for (int mi = 0; mi < 2; mi++)
      #pragma unroll
      for (int ni = 0; ni < 2; ni++) {
        accG[mi][ni] = __builtin_amdgcn_wmma_f32_16x16x32_bf16(
            false, af[mi].v, false, gf[ni].v, (short)0, accG[mi][ni], false, false);
        accU[mi][ni] = __builtin_amdgcn_wmma_f32_16x16x32_bf16(
            false, af[mi].v, false, uf[ni].v, (short)0, accU[mi][ni], false, false);
      }
  }

  int rsel = (lane >> 4) * 8, cc = lane & 15;
  #pragma unroll
  for (int mi = 0; mi < 2; mi++)
    #pragma unroll
    for (int ni = 0; ni < 2; ni++) {
      long col = colBase + wn * 32 + ni * 16 + cc;
      #pragma unroll
      for (int r = 0; r < 8; r++) {
        long row = rowBase + wm * 32 + mi * 16 + rsel + r;
        float g = accG[mi][ni][r], u = accU[mi][ni][r];
        act[row * (long)IM + col] = f2bf(siluf(g) * u);
      }
    }
}

// ---------------- orchestration ----------------
extern "C" void kernel_launch(void* const* d_in, const int* in_sizes, int n_in,
                              void* d_out, int out_size, void* d_ws, size_t ws_size,
                              hipStream_t stream) {
  (void)in_sizes; (void)n_in; (void)out_size; (void)ws_size;
  const float* hidden   = (const float*)d_in[1];
  const float* residual = (const float*)d_in[2];
  const float* w_inproj = (const float*)d_in[3];
  const float* conv_w   = (const float*)d_in[4];
  const float* conv_b   = (const float*)d_in[5];
  const float* A_log    = (const float*)d_in[6];
  const float* Dv       = (const float*)d_in[7];
  const float* dt_bias  = (const float*)d_in[8];
  const float* mixer_w  = (const float*)d_in[9];
  const float* w_outp   = (const float*)d_in[10];
  const float* ln_in_w  = (const float*)d_in[11];
  const float* ln_ff_w  = (const float*)d_in[12];
  const float* w_gu     = (const float*)d_in[13];
  const float* w_dn     = (const float*)d_in[14];

  float* out_f   = (float*)d_out;                          // [ROWS*HID]
  float* resid2  = out_f + (long)ROWS * HID;               // [ROWS*HID]

  char* ws = (char*)d_ws;
  size_t off = 0;
  auto carve = [&](size_t bytes) { size_t r = off; off += (bytes + 255) & ~(size_t)255; return r; };

  unsigned short* bWin  = (unsigned short*)(ws + carve((size_t)DPROJ * HID * 2));
  unsigned short* bWout = (unsigned short*)(ws + carve((size_t)HID * DI * 2));
  unsigned short* bWgu  = (unsigned short*)(ws + carve((size_t)2 * IM * HID * 2));
  unsigned short* bWdn  = (unsigned short*)(ws + carve((size_t)HID * IM * 2));
  unsigned short* hsb   = (unsigned short*)(ws + carve((size_t)ROWS * HID * 2));   // reused as hs2
  float*          proj  = (float*)        (ws + carve((size_t)ROWS * DPROJ * 4)); // reused as act (bf16)
  float*          xbc   = (float*)        (ws + carve((size_t)ROWS * CONV * 4));  // reused as ybf16
  float*          dtv   = (float*)        (ws + carve((size_t)ROWS * NH * 4));
  float*          dAv   = (float*)        (ws + carve((size_t)ROWS * NH * 4));
  float*          yf    = (float*)        (ws + carve((size_t)ROWS * DI * 4));
  unsigned short* ybf   = (unsigned short*)xbc;    // alias: xbc dead after scan
  unsigned short* actb  = (unsigned short*)proj;   // alias: proj dead after gatenorm
  unsigned short* hs2b  = hsb;                     // alias: hs dead after in_proj GEMM

  // 1) weights -> bf16
  cvt_bf16_kernel<<<2048, 256, 0, stream>>>(w_inproj, bWin,  (long)DPROJ * HID);
  cvt_bf16_kernel<<<2048, 256, 0, stream>>>(w_outp,   bWout, (long)HID * DI);
  cvt_bf16_kernel<<<2048, 256, 0, stream>>>(w_gu,     bWgu,  (long)2 * IM * HID);
  cvt_bf16_kernel<<<2048, 256, 0, stream>>>(w_dn,     bWdn,  (long)HID * IM);

  // 2) resid = hidden + residual (stored in d_out residual2 slot); hs = rms(resid)*w -> bf16
  addnorm_kernel<<<ROWS, 256, 0, stream>>>(hidden, residual, ln_in_w, resid2, hsb);

  // 3) proj = hs @ in_proj_w^T   (M=4096, K=2048, E=8512; 8512 = 64*133 -> BN=64)
  wmma_gemm_kernel<<<dim3(DPROJ / BN, ROWS / BM), 256, 0, stream>>>(
      hsb, bWin, proj, nullptr, ROWS, HID, DPROJ);

  // 4) conv + silu; dt / dA
  conv_silu_kernel<<<4096, 256, 0, stream>>>(proj, conv_w, conv_b, xbc);
  dt_kernel<<<(ROWS * NH + 255) / 256, 256, 0, stream>>>(proj, dt_bias, A_log, dtv, dAv);

  // 5) recurrent scan -> y (+ D*x)
  scan_kernel<<<BB * NH, 256, 0, stream>>>(xbc, dtv, dAv, Dv, yf);

  // 6) y*silu(z), mixer rmsnorm -> bf16
  gatenorm_kernel<<<ROWS, 256, 0, stream>>>(yf, proj, mixer_w, ybf);

  // 7) residual2 = y_bf @ out_proj^T + resid   (in place in d_out; E=2048 -> BN=128)
  wmma_gemm128_kernel<<<dim3(HID / BN2, ROWS / BM), 256, 0, stream>>>(
      ybf, bWout, resid2, resid2, ROWS, DI, HID);

  // 8) hs2 = rms(residual2)*w -> bf16
  addnorm_kernel<<<ROWS, 256, 0, stream>>>(resid2, nullptr, ln_ff_w, nullptr, hs2b);

  // 9) act = silu(hs2@Wg^T) * (hs2@Wu^T) -> bf16 (fused dual-tile GEMM)
  wmma_gateup_kernel<<<dim3(IM / BN, ROWS / BM), 256, 0, stream>>>(
      hs2b, bWgu, bWgu + (long)IM * HID, actb, ROWS, HID);

  // 10) out = act @ down^T   (E=2048 -> BN=128)
  wmma_gemm128_kernel<<<dim3(HID / BN2, ROWS / BM), 256, 0, stream>>>(
      actb, bWdn, out_f, nullptr, ROWS, IM, HID);
}
